// MultiHeadAttentionBlock_29961691857757
// MI455X (gfx1250) — compile-verified, matrix operand order checked by reading, not run on hardware
//
#include <hip/hip_runtime.h>

#define DEV __device__ __forceinline__

typedef __bf16 bf16_t;
typedef bf16_t v16bf __attribute__((ext_vector_type(16)));
typedef float  v8f   __attribute__((ext_vector_type(8)));

// Problem dims (fixed by reference)
static constexpr int Bc = 4, Sc = 2048, Dm = 1024, Hc = 16, Dk = 64;
static constexpr int Mrows = Bc * Sc;        // 8192
static constexpr int LDT = 56;               // GEMM LDS row stride (elems): 112B, 16B-aligned, conflict-free
static constexpr int LQ  = 72;               // attn  LDS row stride (elems): 144B, 16B-aligned, conflict-free
static constexpr float SOFT_SCALE = 0.18033688f;  // (1/sqrt(64)) * log2(e)

DEV unsigned short f2bf(float x) {           // f32 -> bf16, round-to-nearest-even
  unsigned int u = __float_as_uint(x);
  u += 0x7FFFu + ((u >> 16) & 1u);
  return (unsigned short)(u >> 16);
}

// ---------------------------------------------------------------------------
// CDNA5 async global->LDS copy (ASYNCcnt path).  Per-lane 16B copy:
//   copies 16B from global vaddr into LDS at vdst   (GLOBAL_LOAD_ASYNC_TO_LDS_B128, §15.18.3 op 98)
// Generic pointers into __shared__ carry the wave-relative LDS offset in
// addr[31:0] (ISA §10.2 aperture mapping), so the low dword is the LDS vdst.
// ---------------------------------------------------------------------------
#if defined(__gfx1250__)
DEV void cp_async_b128(void* lds_dst, const void* gsrc) {
  unsigned       ldsa = (unsigned)(unsigned long long)lds_dst;
  unsigned long long ga = (unsigned long long)gsrc;
  asm volatile("global_load_async_to_lds_b128 %0, %1, off"
               :: "v"(ldsa), "v"(ga)
               : "memory");
}
DEV void wait_async() {
  asm volatile("s_wait_asynccnt 0x0" ::: "memory");
}
#else
DEV void cp_async_b128(void* lds_dst, const void* gsrc) {
  *(uint4*)lds_dst = *(const uint4*)gsrc;
}
DEV void wait_async() {}
#endif

// Load one 16x32 bf16 fragment (A- or B-layout) from a row-major tile.
// Per ISA 7.12.2: lane L -> row (L&15); K-chunks at (L>>4)*8 and (L>>4)*8+16.
DEV v16bf ld_frag(const unsigned short* base, int stride, int lane) {
  const int r = lane & 15, h = lane >> 4;
  const unsigned short* p = base + r * stride + h * 8;
  union { uint4 u[2]; v16bf v; } un;
  un.u[0] = *(const uint4*)(p);
  un.u[1] = *(const uint4*)(p + 16);
  return un.v;
}

DEV v8f wmma_bf16(v16bf a, v16bf b, v8f c) {
  return __builtin_amdgcn_wmma_f32_16x16x32_bf16(false, a, false, b, (short)0, c, false, false);
}

// ---------------------------------------------------------------------------
// Kernel 1: fused QKV projections.  grid = (M/128, N/128, 3); block = 256.
// C[m,n] = X[m,:] . W[n,:] + bias[n], written bf16 into head-major layouts:
//   z=0 -> Q [B,H,S,Dk], z=1 -> K [B,H,S,Dk], z=2 -> V^T [B,H,Dk,S]
// ---------------------------------------------------------------------------
__global__ __launch_bounds__(256)
void qkv_proj_kernel(const float* __restrict__ xq, const float* __restrict__ xk,
                     const float* __restrict__ xv,
                     const float* __restrict__ wq, const float* __restrict__ wk,
                     const float* __restrict__ wv,
                     const float* __restrict__ bq, const float* __restrict__ bk,
                     const float* __restrict__ bv,
                     unsigned short* __restrict__ Qb,
                     unsigned short* __restrict__ Kb,
                     unsigned short* __restrict__ Vb) {
  __shared__ unsigned short As[128 * LDT];
  __shared__ unsigned short Bs[128 * LDT];

  const int tid = threadIdx.x, lane = tid & 31, wid = tid >> 5;
  const int z = blockIdx.z;
  const float* X  = (z == 0) ? xq : (z == 1) ? xk : xv;
  const float* W  = (z == 0) ? wq : (z == 1) ? wk : wv;
  const float* Bi = (z == 0) ? bq : (z == 1) ? bk : bv;
  unsigned short* Out = (z == 0) ? Qb : (z == 1) ? Kb : Vb;

  const int m0 = blockIdx.x * 128;
  const int n0 = blockIdx.y * 128;
  const int wm = (wid >> 2) * 64;    // 2 row-waves
  const int wn = (wid & 3) * 32;     // 4 col-waves

  v8f acc[4][2];
  const v8f zero8 = {0.f, 0.f, 0.f, 0.f, 0.f, 0.f, 0.f, 0.f};
#pragma unroll
  for (int i = 0; i < 4; ++i)
#pragma unroll
    for (int j = 0; j < 2; ++j) acc[i][j] = zero8;

  const int lr = tid >> 3;           // 0..31
  const int lc = (tid & 7) * 4;      // 0..28

  for (int kt = 0; kt < Dm; kt += 32) {
    __syncthreads();
#pragma unroll
    for (int i = 0; i < 4; ++i) {
      const int row = lr + i * 32;
      float4 a = *(const float4*)(X + (size_t)(m0 + row) * Dm + kt + lc);
      unsigned short* da = &As[row * LDT + lc];
      da[0] = f2bf(a.x); da[1] = f2bf(a.y); da[2] = f2bf(a.z); da[3] = f2bf(a.w);
      float4 w = *(const float4*)(W + (size_t)(n0 + row) * Dm + kt + lc);
      unsigned short* db = &Bs[row * LDT + lc];
      db[0] = f2bf(w.x); db[1] = f2bf(w.y); db[2] = f2bf(w.z); db[3] = f2bf(w.w);
    }
    __syncthreads();

    v16bf af[4], bfrag[2];
#pragma unroll
    for (int i = 0; i < 4; ++i) af[i] = ld_frag(&As[(wm + i * 16) * LDT], LDT, lane);
#pragma unroll
    for (int j = 0; j < 2; ++j) bfrag[j] = ld_frag(&Bs[(wn + j * 16) * LDT], LDT, lane);
#pragma unroll
    for (int i = 0; i < 4; ++i)
#pragma unroll
      for (int j = 0; j < 2; ++j) acc[i][j] = wmma_bf16(af[i], bfrag[j], acc[i][j]);
  }

  // Epilogue: C-fragment (VGPR r, lane L) -> row r+8*(L>>4), col (L&15)
  const int nl = lane & 15, half = lane >> 4;
#pragma unroll
  for (int j = 0; j < 2; ++j) {
    const int gn = n0 + wn + j * 16 + nl;
    const float bias = Bi[gn];
    const int h = gn >> 6, d = gn & 63;
#pragma unroll
    for (int i = 0; i < 4; ++i) {
#pragma unroll
      for (int r = 0; r < 8; ++r) {
        const int gm = m0 + wm + i * 16 + r + half * 8;
        const int b = gm >> 11, s = gm & (Sc - 1);
        const float val = acc[i][j][r] + bias;
        size_t off;
        if (z == 2) off = (((size_t)(b * Hc + h)) * Dk + d) * Sc + s;   // V^T
        else        off = (((size_t)(b * Hc + h)) * Sc + s) * Dk + d;   // Q,K
        Out[off] = f2bf(val);
      }
    }
  }
}

// ---------------------------------------------------------------------------
// Kernel 2: flash attention.  grid = (S/128, B*H); block = 256 (8 waves x 16 q-rows).
// ---------------------------------------------------------------------------
__global__ __launch_bounds__(256)
void attn_kernel(const unsigned short* __restrict__ Qb,
                 const unsigned short* __restrict__ Kb,
                 const unsigned short* __restrict__ Vb,
                 const int* __restrict__ mask,
                 unsigned short* __restrict__ Ab) {
  __shared__ unsigned short Qs[128 * LQ];    // Q tile (q x d), resident
  __shared__ unsigned short KPs[128 * LQ];   // K tile (key x d, rows 0..63)  UNION  P tile (q x key)
  __shared__ unsigned short Vts[64 * LQ];    // V^T tile (d x key)

  const int tid = threadIdx.x, lane = tid & 31, wid = tid >> 5;
  const int q0 = blockIdx.x * 128;
  const int bh = blockIdx.y;
  const int b = bh >> 4, h = bh & 15;
  const int wrow = wid * 16;
  const int nl = lane & 15, half = lane >> 4;

  const size_t qkBase = (size_t)bh * Sc * Dk;
  const size_t vBase  = (size_t)bh * Dk * Sc;

  // Stage Q tile once (async: pure bf16 copy, no VGPR round-trip)
  {
    const int row = tid >> 3, c = (tid & 7) * 8;
#pragma unroll
    for (int i = 0; i < 4; ++i) {
      const int rr = row + i * 32;
      cp_async_b128(&Qs[rr * LQ + c], Qb + qkBase + (size_t)(q0 + rr) * Dk + c);
    }
  }
  wait_async();
  __syncthreads();

  v16bf qf[2];
  qf[0] = ld_frag(&Qs[wrow * LQ], LQ, lane);
  qf[1] = ld_frag(&Qs[wrow * LQ + 32], LQ, lane);

  const v8f zero8 = {0.f, 0.f, 0.f, 0.f, 0.f, 0.f, 0.f, 0.f};
  v8f of[4];
#pragma unroll
  for (int f = 0; f < 4; ++f) of[f] = zero8;
  float mrun[8], lrun[8];
#pragma unroll
  for (int r = 0; r < 8; ++r) { mrun[r] = -3.0e38f; lrun[r] = 0.f; }

  for (int kt = 0; kt < Sc; kt += 64) {
    __syncthreads();  // previous iteration's P reads done before overwriting KPs/Vts
    {
      const int row = tid >> 3, c = (tid & 7) * 8;
#pragma unroll
      for (int i = 0; i < 2; ++i) {
        const int rr = row + i * 32;
        cp_async_b128(&KPs[rr * LQ + c], Kb + qkBase + (size_t)(kt + rr) * Dk + c);
        cp_async_b128(&Vts[rr * LQ + c], Vb + vBase + (size_t)rr * Sc + kt + c);
      }
    }
    wait_async();
    __syncthreads();

    // S = Q . K^T (B-fragment lane = key row of K tile, contiguous over d)
    v8f sc[4];
#pragma unroll
    for (int j = 0; j < 4; ++j) {
      v16bf k0 = ld_frag(&KPs[(j * 16) * LQ], LQ, lane);
      v16bf k1 = ld_frag(&KPs[(j * 16) * LQ + 32], LQ, lane);
      sc[j] = wmma_bf16(qf[1], k1, wmma_bf16(qf[0], k0, zero8));
    }

    // mask + scale into log2-domain
#pragma unroll
    for (int j = 0; j < 4; ++j) {
      const int gk = kt + j * 16 + nl;
#pragma unroll
      for (int r = 0; r < 8; ++r) {
        const int gq = q0 + wrow + r + half * 8;
        const int mv = mask[(size_t)gq * Sc + gk];
        const float t = sc[j][r] * SOFT_SCALE;
        sc[j][r] = mv ? t : -3.0e8f;
      }
    }

    // online softmax: reduce across the 16-lane N group of each row
#pragma unroll
    for (int r = 0; r < 8; ++r) {
      float mx = fmaxf(fmaxf(sc[0][r], sc[1][r]), fmaxf(sc[2][r], sc[3][r]));
      mx = fmaxf(mx, __shfl_xor(mx, 1, 32));
      mx = fmaxf(mx, __shfl_xor(mx, 2, 32));
      mx = fmaxf(mx, __shfl_xor(mx, 4, 32));
      mx = fmaxf(mx, __shfl_xor(mx, 8, 32));
      const float mnew = fmaxf(mrun[r], mx);
      const float alpha = exp2f(mrun[r] - mnew);
      mrun[r] = mnew;
      float rs = 0.f;
#pragma unroll
      for (int j = 0; j < 4; ++j) {
        const float p = exp2f(sc[j][r] - mnew);
        sc[j][r] = p;
        rs += p;
      }
      rs += __shfl_xor(rs, 1, 32);
      rs += __shfl_xor(rs, 2, 32);
      rs += __shfl_xor(rs, 4, 32);
      rs += __shfl_xor(rs, 8, 32);
      lrun[r] = lrun[r] * alpha + rs;
#pragma unroll
      for (int f = 0; f < 4; ++f) of[f][r] *= alpha;
    }

    __syncthreads();  // all waves finished reading K tile
    // spill P (bf16) into the K/P union region, A-layout source (q-row major)
#pragma unroll
    for (int j = 0; j < 4; ++j)
#pragma unroll
      for (int r = 0; r < 8; ++r)
        KPs[(wrow + r + half * 8) * LQ + j * 16 + nl] = f2bf(sc[j][r]);
    __syncthreads();

    // O += P . V  (B-fragment lane = d column, contiguous over keys in V^T)
    v16bf p0 = ld_frag(&KPs[wrow * LQ], LQ, lane);
    v16bf p1 = ld_frag(&KPs[wrow * LQ + 32], LQ, lane);
#pragma unroll
    for (int f = 0; f < 4; ++f) {
      v16bf v0 = ld_frag(&Vts[(f * 16) * LQ], LQ, lane);
      v16bf v1 = ld_frag(&Vts[(f * 16) * LQ + 32], LQ, lane);
      of[f] = wmma_bf16(p1, v1, wmma_bf16(p0, v0, of[f]));
    }
  }

  // normalize + write attn output [B,S,Dm] bf16
#pragma unroll
  for (int r = 0; r < 8; ++r) {
    const float inv = 1.0f / lrun[r];
    const int gq = q0 + wrow + r + half * 8;
    const size_t rowOff = ((size_t)b * Sc + gq) * Dm + h * Dk;
#pragma unroll
    for (int f = 0; f < 4; ++f) Ab[rowOff + f * 16 + nl] = f2bf(of[f][r] * inv);
  }
}

// ---------------------------------------------------------------------------
// Kernel 3: output projection (bf16 A from workspace, f32 W, f32 out + bias)
// ---------------------------------------------------------------------------
__global__ __launch_bounds__(256)
void out_proj_kernel(const unsigned short* __restrict__ Ain,
                     const float* __restrict__ Wo, const float* __restrict__ bo,
                     float* __restrict__ Out) {
  __shared__ unsigned short As[128 * LDT];
  __shared__ unsigned short Bs[128 * LDT];

  const int tid = threadIdx.x, lane = tid & 31, wid = tid >> 5;
  const int m0 = blockIdx.x * 128;
  const int n0 = blockIdx.y * 128;
  const int wm = (wid >> 2) * 64;
  const int wn = (wid & 3) * 32;

  v8f acc[4][2];
  const v8f zero8 = {0.f, 0.f, 0.f, 0.f, 0.f, 0.f, 0.f, 0.f};
#pragma unroll
  for (int i = 0; i < 4; ++i)
#pragma unroll
    for (int j = 0; j < 2; ++j) acc[i][j] = zero8;

  const int ar = tid >> 2, ac = (tid & 3) * 8;  // bf16 A loader (async copy)
  const int lr = tid >> 3, lc = (tid & 7) * 4;  // f32 W loader (converting)

  for (int kt = 0; kt < Dm; kt += 32) {
    __syncthreads();
#pragma unroll
    for (int i = 0; i < 2; ++i) {
      const int rr = ar + i * 64;
      cp_async_b128(&As[rr * LDT + ac], Ain + (size_t)(m0 + rr) * Dm + kt + ac);
    }
#pragma unroll
    for (int i = 0; i < 4; ++i) {
      const int row = lr + i * 32;
      float4 w = *(const float4*)(Wo + (size_t)(n0 + row) * Dm + kt + lc);
      unsigned short* db = &Bs[row * LDT + lc];
      db[0] = f2bf(w.x); db[1] = f2bf(w.y); db[2] = f2bf(w.z); db[3] = f2bf(w.w);
    }
    wait_async();
    __syncthreads();

    v16bf af[4], bfrag[2];
#pragma unroll
    for (int i = 0; i < 4; ++i) af[i] = ld_frag(&As[(wm + i * 16) * LDT], LDT, lane);
#pragma unroll
    for (int j = 0; j < 2; ++j) bfrag[j] = ld_frag(&Bs[(wn + j * 16) * LDT], LDT, lane);
#pragma unroll
    for (int i = 0; i < 4; ++i)
#pragma unroll
      for (int j = 0; j < 2; ++j) acc[i][j] = wmma_bf16(af[i], bfrag[j], acc[i][j]);
  }

  const int nl = lane & 15, half = lane >> 4;
#pragma unroll
  for (int j = 0; j < 2; ++j) {
    const int gn = n0 + wn + j * 16 + nl;
    const float bias = bo[gn];
#pragma unroll
    for (int i = 0; i < 4; ++i)
#pragma unroll
      for (int r = 0; r < 8; ++r) {
        const int gm = m0 + wm + i * 16 + r + half * 8;
        Out[(size_t)gm * Dm + gn] = acc[i][j][r] + bias;
      }
  }
}

// ---------------------------------------------------------------------------
// Workspace layout (bf16/ushort): Q | K | V^T | attn-out = 4 * 8192*1024 * 2B = 67.1 MB
// ---------------------------------------------------------------------------
extern "C" void kernel_launch(void* const* d_in, const int* in_sizes, int n_in,
                              void* d_out, int out_size, void* d_ws, size_t ws_size,
                              hipStream_t stream) {
  const float* q  = (const float*)d_in[0];
  const float* k  = (const float*)d_in[1];
  const float* v  = (const float*)d_in[2];
  const int* mask = (const int*)d_in[3];
  const float* wq = (const float*)d_in[4];
  const float* bq = (const float*)d_in[5];
  const float* wk = (const float*)d_in[6];
  const float* bk = (const float*)d_in[7];
  const float* wv = (const float*)d_in[8];
  const float* bv = (const float*)d_in[9];
  const float* wo = (const float*)d_in[10];
  const float* bo = (const float*)d_in[11];

  const size_t plane = (size_t)Mrows * Dm;  // 8192*1024 elems
  unsigned short* Qb = (unsigned short*)d_ws;
  unsigned short* Kb = Qb + plane;
  unsigned short* Vb = Kb + plane;
  unsigned short* Ab = Vb + plane;

  qkv_proj_kernel<<<dim3(Mrows / 128, Dm / 128, 3), 256, 0, stream>>>(
      q, k, v, wq, wk, wv, bq, bk, bv, Qb, Kb, Vb);
  attn_kernel<<<dim3(Sc / 128, Bc * Hc), 256, 0, stream>>>(Qb, Kb, Vb, mask, Ab);
  out_proj_kernel<<<dim3(Mrows / 128, Dm / 128), 256, 0, stream>>>(Ab, wo, bo,
                                                                   (float*)d_out);
}
